// CausalSelfAttention_19928648254227
// MI455X (gfx1250) — matrix operand order repaired and checked by reading, not compiled, not run on hardware
//
#include <hip/hip_runtime.h>

// ---------------------------------------------------------------------------
// MLA causal attention for gfx1250 (MI455X), bf16 WMMA everywhere.
// GEMM tiles staged to LDS via CDNA5 async global->LDS copies when available.
// ---------------------------------------------------------------------------

typedef unsigned short u16;
typedef __attribute__((ext_vector_type(8)))  u16    u16x8;
typedef __attribute__((ext_vector_type(16))) __bf16 v16bf;
typedef __attribute__((ext_vector_type(8)))  float  v8f;

#define B_    4
#define T_    1024
#define C_    2048
#define NH_   16
#define HS_   128
#define NLQ_  1536
#define NLKV_ 512
#define DHR_  64

// CDNA5 async global->LDS copy path (ASYNCcnt-tracked), guarded so the file
// still compiles on toolchains without the builtin.
#if defined(__has_builtin)
#if __has_builtin(__builtin_amdgcn_global_load_async_to_lds_b128) && \
    __has_builtin(__builtin_amdgcn_s_wait_asynccnt)
#define USE_ASYNC_LDS 1
#endif
#endif
#ifndef USE_ASYNC_LDS
#define USE_ASYNC_LDS 0
#endif

#if USE_ASYNC_LDS
// Builtin signature (from hipcc diagnostic): param 0 is
// 'int __attribute__((vector_size(16)))*' in the global address space.
typedef int v4i_ __attribute__((vector_size(16)));
typedef __attribute__((address_space(1))) v4i_* as1_v4i;
typedef __attribute__((address_space(3))) v4i_* as3_v4i;

__device__ __forceinline__ void async_cp16(const u16* g, u16* l) {
  __builtin_amdgcn_global_load_async_to_lds_b128((as1_v4i)g, (as3_v4i)l, 0, 0);
}
#endif

__device__ __forceinline__ u16 f2bf(float f) {
  unsigned u = __float_as_uint(f);
  u += 0x7FFFu + ((u >> 16) & 1u);            // RNE truncate fp32 -> bf16
  return (u16)(u >> 16);
}

// A-fragment (16x32, MxK): lane m holds row m; hl=0 -> K 0..7 & 16..23,
// hl=1 -> K 8..15 & 24..31  (ISA 7.12.2 16-bit A layout)
__device__ __forceinline__ v16bf frag_a(const u16* p, int hl) {
  union { u16x8 u[2]; v16bf v; } t;
  t.u[0] = *(const u16x8*)(p + hl * 8);
  t.u[1] = *(const u16x8*)(p + 16 + hl * 8);
  return t.v;
}
// B-fragment (32x16, KxN) from NT storage (row n, K contiguous):
// lane n holds col n; hl=0 -> K 0..7 & 16..23, hl=1 -> K 8..15 & 24..31
__device__ __forceinline__ v16bf frag_b(const u16* p, int hl) {
  union { u16x8 u[2]; v16bf v; } t;
  t.u[0] = *(const u16x8*)(p + hl * 8);
  t.u[1] = *(const u16x8*)(p + 16 + hl * 8);
  return t.v;
}

__device__ __forceinline__ v8f wmma_bf(v16bf a, v16bf b, v8f c) {
  return __builtin_amdgcn_wmma_f32_16x16x32_bf16(false, a, false, b,
                                                 (short)0, c, false, false);
}

__device__ __forceinline__ float rmax16(float v) {
#pragma unroll
  for (int m = 1; m < 16; m <<= 1) v = fmaxf(v, __shfl_xor(v, m, 16));
  return v;
}
__device__ __forceinline__ float rsum16(float v) {
#pragma unroll
  for (int m = 1; m < 16; m <<= 1) v += __shfl_xor(v, m, 16);
  return v;
}

// ---------------------------------------------------------------------------
// Elementwise / transpose helpers
// ---------------------------------------------------------------------------
__global__ void cvt_kernel(const float* __restrict__ in, u16* __restrict__ out, long n) {
  long i = (long)blockIdx.x * blockDim.x + threadIdx.x;
  if (i < n) out[i] = f2bf(in[i]);
}

// out[c*R + r] = in[r*Cc + c], batched over blockIdx.y
__global__ void tconv_kernel(const float* __restrict__ in, u16* __restrict__ out,
                             int R, int Cc, long inb, long outb) {
  long i = (long)blockIdx.x * blockDim.x + threadIdx.x;
  if (i >= (long)R * Cc) return;
  in  += (long)blockIdx.y * inb;
  out += (long)blockIdx.y * outb;
  int rr = (int)(i / Cc);
  int cc = (int)(i - (long)rr * Cc);
  out[(long)cc * R + rr] = f2bf(in[i]);
}

// c_kv (B,T,512) bf16 -> c_kvT (B,512,T) bf16
__global__ void tbf_kernel(const u16* __restrict__ in, u16* __restrict__ out) {
  long i = (long)blockIdx.x * blockDim.x + threadIdx.x;
  if (i >= (long)B_ * T_ * NLKV_) return;
  int c = (int)(i & (NLKV_ - 1));
  long bt = i >> 9;
  int t = (int)(bt & (T_ - 1));
  int b = (int)(bt >> 10);
  out[((long)b * NLKV_ + c) * T_ + t] = in[i];
}

__global__ void rope_k_kernel(const float* __restrict__ ckr, const float* __restrict__ cosb,
                              const float* __restrict__ sinb, u16* __restrict__ kr) {
  long idx = (long)blockIdx.x * blockDim.x + threadIdx.x;   // B*T*32 pairs
  if (idx >= (long)B_ * T_ * 32) return;
  int j = (int)(idx & 31);
  long bt = idx >> 5;
  int t = (int)(bt & (T_ - 1));
  float re = ckr[bt * 64 + 2 * j];
  float im = ckr[bt * 64 + 2 * j + 1];
  float c = cosb[t * 32 + j], s = sinb[t * 32 + j];
  kr[bt * 64 + 2 * j]     = f2bf(re * c - im * s);
  kr[bt * 64 + 2 * j + 1] = f2bf(re * s + im * c);
}

// c_qr (B,T,NH,64) f32 -> q_r (B,NH,T,64) bf16 with RoPE
__global__ void rope_q_kernel(const float* __restrict__ cqr, const float* __restrict__ cosb,
                              const float* __restrict__ sinb, u16* __restrict__ qr) {
  long idx = (long)blockIdx.x * blockDim.x + threadIdx.x;   // B*T*NH*32
  if (idx >= (long)B_ * T_ * NH_ * 32) return;
  int j = (int)(idx & 31);
  long rest = idx >> 5;
  int h = (int)(rest & (NH_ - 1)); rest >>= 4;
  int t = (int)(rest & (T_ - 1));
  int b = (int)(rest >> 10);
  long in  = (((long)(b * T_ + t) * NH_) + h) * 64;
  long out = (((long)(b * NH_ + h) * T_) + t) * 64;
  float re = cqr[in + 2 * j], im = cqr[in + 2 * j + 1];
  float c = cosb[t * 32 + j], s = sinb[t * 32 + j];
  qr[out + 2 * j]     = f2bf(re * c - im * s);
  qr[out + 2 * j + 1] = f2bf(re * s + im * c);
}

// ---------------------------------------------------------------------------
// Generic batched NT GEMM:  D[m,n] = sum_k A[m,k] * Bm[n,k]
// Tile 128x64x32, 256 threads = 8 waves, each wave 32x32 (2x2 WMMA).
// z = z1*nz2 + z2 ; per-operand flat element offsets per z1/z2.
// Tiles staged to LDS via async global->LDS (ASYNCcnt) when available.
// ---------------------------------------------------------------------------
#define BM 128
#define BN 64
#define BK 32
#define LS 40   // padded LDS row stride in halves (32 + 8)

template <bool F32OUT>
__global__ __launch_bounds__(256) void gemm_nt(
    const u16* __restrict__ A, int lda,
    const u16* __restrict__ Bm, int ldb,
    void* __restrict__ Dv, int ldd,
    int nz2, long aoz1, long aoz2, long boz1, long boz2, long doz1, long doz2,
    int K) {
  __shared__ u16 sA[BM * LS];
  __shared__ u16 sB[BN * LS];
  const int z = blockIdx.z;
  const int z1 = z / nz2, z2 = z - z1 * nz2;
  A  += z1 * aoz1 + z2 * aoz2;
  Bm += z1 * boz1 + z2 * boz2;
  const long doff = z1 * doz1 + z2 * doz2;
  const int mBase = blockIdx.y * BM;
  const int nBase = blockIdx.x * BN;
  const int tid = threadIdx.x;
  const int lane = tid & 31, wave = tid >> 5;
  const int hl = lane >> 4, r = lane & 15;
  const int wm = wave & 3, wn = wave >> 2;
  v8f acc[2][2] = {};

  const int aRow0 = tid >> 2,         aCh0 = (tid & 3) * 8;
  const int aRow1 = (tid + 256) >> 2, aCh1 = ((tid + 256) & 3) * 8;

  for (int k0 = 0; k0 < K; k0 += BK) {
#if USE_ASYNC_LDS
    // CDNA5 path: L2 -> LDS without a VGPR round trip; tracked on ASYNCcnt.
    async_cp16(A  + (long)(mBase + aRow0) * lda + k0 + aCh0, sA + aRow0 * LS + aCh0);
    async_cp16(A  + (long)(mBase + aRow1) * lda + k0 + aCh1, sA + aRow1 * LS + aCh1);
    async_cp16(Bm + (long)(nBase + aRow0) * ldb + k0 + aCh0, sB + aRow0 * LS + aCh0);
    __builtin_amdgcn_s_wait_asynccnt(0);
    __syncthreads();
#else
    u16x8 va0 = *(const u16x8*)(A + (long)(mBase + aRow0) * lda + k0 + aCh0);
    u16x8 va1 = *(const u16x8*)(A + (long)(mBase + aRow1) * lda + k0 + aCh1);
    u16x8 vb0 = *(const u16x8*)(Bm + (long)(nBase + aRow0) * ldb + k0 + aCh0);
    *(u16x8*)(sA + aRow0 * LS + aCh0) = va0;
    *(u16x8*)(sA + aRow1 * LS + aCh1) = va1;
    *(u16x8*)(sB + aRow0 * LS + aCh0) = vb0;
    __syncthreads();
#endif
    v16bf a0 = frag_a(sA + (wm * 32 + r) * LS, hl);
    v16bf a1 = frag_a(sA + (wm * 32 + 16 + r) * LS, hl);
    v16bf b0 = frag_b(sB + (wn * 32 + r) * LS, hl);
    v16bf b1 = frag_b(sB + (wn * 32 + 16 + r) * LS, hl);
    acc[0][0] = wmma_bf(a0, b0, acc[0][0]);
    acc[0][1] = wmma_bf(a0, b1, acc[0][1]);
    acc[1][0] = wmma_bf(a1, b0, acc[1][0]);
    acc[1][1] = wmma_bf(a1, b1, acc[1][1]);
    __syncthreads();
  }

  float* Df = (float*)Dv;
  u16*   Db = (u16*)Dv;
#pragma unroll
  for (int mt = 0; mt < 2; ++mt)
#pragma unroll
    for (int nt = 0; nt < 2; ++nt)
#pragma unroll
      for (int i = 0; i < 8; ++i) {
        long row = mBase + wm * 32 + mt * 16 + i + 8 * hl;
        long col = nBase + wn * 32 + nt * 16 + r;
        long idx = doff + row * (long)ldd + col;
        if (F32OUT) Df[idx] = acc[mt][nt][i];
        else        Db[idx] = f2bf(acc[mt][nt][i]);
      }
}

// ---------------------------------------------------------------------------
// Flash attention over latent space.
// Q feats = [q_abs(512) | q_r(64)], K feats = [c_kv(512) | k_r(64)], V = c_kv.
// Block = 4 waves, 64 query rows, one (b,h), one 128-col V chunk (grid.y).
// Key blocks of 32, causal skip. y_lat (B,NH,T,512) bf16.
// ---------------------------------------------------------------------------
#define PS 40

__global__ __launch_bounds__(128) void flash_mla(
    const u16* __restrict__ qabs, const u16* __restrict__ qr,
    const u16* __restrict__ ckv,  const u16* __restrict__ kr,
    const u16* __restrict__ ckvT, u16* __restrict__ ylat) {
  __shared__ u16 sP[4 * 16 * PS];
  const int lane = threadIdx.x & 31, wave = threadIdx.x >> 5;
  const int hl = lane >> 4, r = lane & 15;
  const int bh = blockIdx.z, b = bh >> 4;
  const int m0 = blockIdx.x * 64 + wave * 16;    // this wave's 16 query rows
  const int vbase = blockIdx.y * 128;
  const int nk = blockIdx.x * 2 + 2;             // causal key-block count

  const u16* Qa = qabs + ((long)bh * T_ + m0) * 512;
  const u16* Qr = qr   + ((long)bh * T_ + m0) * 64;
  const u16* Ka = ckv  + (long)b * T_ * 512;
  const u16* Kr = kr   + (long)b * T_ * 64;
  const u16* Vt = ckvT + ((long)b * NLKV_ + vbase) * T_;
  const float scale = 0.0721687836487032f;       // 1/sqrt(192)

  float m_i[8], l_i[8];
  v8f O[8] = {};
#pragma unroll
  for (int i = 0; i < 8; ++i) { m_i[i] = -1e30f; l_i[i] = 0.f; }
  u16* myP = sP + wave * 16 * PS;

  for (int sb = 0; sb < nk; ++sb) {
    const int s0 = sb * 32;
    v8f S0 = {}, S1 = {};
#pragma unroll 4
    for (int kk = 0; kk < 16; ++kk) {            // 512 absorbed features
      v16bf a  = frag_a(Qa + (long)r * 512 + kk * 32, hl);
      v16bf k0 = frag_b(Ka + (long)(s0 + r) * 512 + kk * 32, hl);
      v16bf k1 = frag_b(Ka + (long)(s0 + 16 + r) * 512 + kk * 32, hl);
      S0 = wmma_bf(a, k0, S0);
      S1 = wmma_bf(a, k1, S1);
    }
#pragma unroll
    for (int kk = 0; kk < 2; ++kk) {             // 64 RoPE features
      v16bf a  = frag_a(Qr + (long)r * 64 + kk * 32, hl);
      v16bf k0 = frag_b(Kr + (long)(s0 + r) * 64 + kk * 32, hl);
      v16bf k1 = frag_b(Kr + (long)(s0 + 16 + r) * 64 + kk * 32, hl);
      S0 = wmma_bf(a, k0, S0);
      S1 = wmma_bf(a, k1, S1);
    }
    float al[8];
#pragma unroll
    for (int i = 0; i < 8; ++i) {
      const int t = m0 + i + 8 * hl;
      float s0v = (s0 + r      <= t) ? S0[i] * scale : -1e30f;
      float s1v = (s0 + 16 + r <= t) ? S1[i] * scale : -1e30f;
      float mnew = fmaxf(m_i[i], rmax16(fmaxf(s0v, s1v)));
      float a  = __expf(m_i[i] - mnew);
      float p0 = __expf(s0v - mnew);
      float p1 = __expf(s1v - mnew);
      l_i[i] = l_i[i] * a + rsum16(p0 + p1);
      m_i[i] = mnew;
      al[i] = a;
      myP[(i + 8 * hl) * PS + r]      = f2bf(p0);
      myP[(i + 8 * hl) * PS + 16 + r] = f2bf(p1);
    }
#pragma unroll
    for (int nt = 0; nt < 8; ++nt)
#pragma unroll
      for (int i = 0; i < 8; ++i) O[nt][i] *= al[i];
    __syncthreads();
    v16bf pf = frag_a(myP + r * PS, hl);         // P as A-fragment (16x32)
#pragma unroll
    for (int nt = 0; nt < 8; ++nt) {
      v16bf vf = frag_b(Vt + (long)(nt * 16 + r) * T_ + s0, hl);
      O[nt] = wmma_bf(pf, vf, O[nt]);
    }
    __syncthreads();
  }
#pragma unroll
  for (int i = 0; i < 8; ++i) {
    const float inv = 1.f / l_i[i];
    const long t = m0 + i + 8 * hl;
#pragma unroll
    for (int nt = 0; nt < 8; ++nt)
      ylat[((long)bh * T_ + t) * 512 + vbase + nt * 16 + r] = f2bf(O[nt][i] * inv);
  }
}

// ---------------------------------------------------------------------------
// Host launch
// ---------------------------------------------------------------------------
extern "C" void kernel_launch(void* const* d_in, const int* in_sizes, int n_in,
                              void* d_out, int out_size, void* d_ws, size_t ws_size,
                              hipStream_t stream) {
  const float* x    = (const float*)d_in[0];
  const float* cosb = (const float*)d_in[1];
  const float* sinb = (const float*)d_in[2];
  const float* Wdq  = (const float*)d_in[3];
  const float* Wuq  = (const float*)d_in[4];
  const float* Wdkv = (const float*)d_in[5];
  const float* Wuk  = (const float*)d_in[6];
  const float* Wuv  = (const float*)d_in[7];
  const float* Wqr  = (const float*)d_in[8];
  const float* Wkr  = (const float*)d_in[9];
  const float* Wo   = (const float*)d_in[10];
  float* y = (float*)d_out;

  char* ws = (char*)d_ws;
  size_t off = 0;
  auto alloc = [&](size_t bytes) -> void* {
    off = (off + 255) & ~(size_t)255;
    void* p = ws + off;
    off += bytes;
    return p;
  };
  const long nBT = (long)B_ * T_;
  u16*   x_bf    = (u16*)alloc(nBT * C_ * 2);
  u16*   wdq_bf  = (u16*)alloc((long)NLQ_ * C_ * 2);
  u16*   wdkv_bf = (u16*)alloc((long)NLKV_ * C_ * 2);
  u16*   wkr_bf  = (u16*)alloc((long)DHR_ * C_ * 2);
  u16*   wqr_bf  = (u16*)alloc((long)NH_ * DHR_ * NLQ_ * 2);
  u16*   wuq_bf  = (u16*)alloc((long)C_ * NLQ_ * 2);
  u16*   wo_bf   = (u16*)alloc((long)C_ * C_ * 2);
  u16*   wuvT    = (u16*)alloc((long)NLKV_ * C_ * 2);
  u16*   wukT    = (u16*)alloc((long)NH_ * NLKV_ * HS_ * 2);
  u16*   cq_bf   = (u16*)alloc(nBT * NLQ_ * 2);
  u16*   ckv     = (u16*)alloc(nBT * NLKV_ * 2);
  u16*   ckvT    = (u16*)alloc(nBT * NLKV_ * 2);
  float* ckr     = (float*)alloc(nBT * DHR_ * 4);
  u16*   kr      = (u16*)alloc(nBT * DHR_ * 2);
  float* cqr     = (float*)alloc(nBT * NH_ * DHR_ * 4);
  u16*   qr      = (u16*)alloc(nBT * NH_ * DHR_ * 2);
  u16*   keffT   = (u16*)alloc((long)NH_ * NLKV_ * NLQ_ * 2);
  u16*   veff    = (u16*)alloc((long)C_ * NLKV_ * 2);
  u16*   qabs    = (u16*)alloc((long)B_ * NH_ * T_ * NLKV_ * 2);
  u16*   ylat    = (u16*)alloc((long)B_ * NH_ * T_ * NLKV_ * 2);   // total ~257 MB

  auto cvt = [&](const float* in, u16* out, long n) {
    cvt_kernel<<<dim3((unsigned)((n + 255) / 256)), dim3(256), 0, stream>>>(in, out, n);
  };
  cvt(x, x_bf, nBT * C_);
  cvt(Wdq, wdq_bf, (long)NLQ_ * C_);
  cvt(Wdkv, wdkv_bf, (long)NLKV_ * C_);
  cvt(Wkr, wkr_bf, (long)DHR_ * C_);
  cvt(Wqr, wqr_bf, (long)NH_ * DHR_ * NLQ_);
  cvt(Wuq, wuq_bf, (long)C_ * NLQ_);
  cvt(Wo, wo_bf, (long)C_ * C_);
  // W_uv (C x NLKV) -> (NLKV x C)
  tconv_kernel<<<dim3((C_ * NLKV_ + 255) / 256, 1), 256, 0, stream>>>(
      Wuv, wuvT, C_, NLKV_, 0, 0);
  // W_uk viewed (NH, HS, NLKV) -> (NH, NLKV, HS)
  tconv_kernel<<<dim3((HS_ * NLKV_ + 255) / 256, NH_), 256, 0, stream>>>(
      Wuk, wukT, HS_, NLKV_, (long)HS_ * NLKV_, (long)HS_ * NLKV_);

  auto gemm = [&](const u16* A, int lda, const u16* Bp, int ldb, void* D, int ldd,
                  int M, int N, int K, bool f32out, int nz1, int nz2,
                  long aoz1, long aoz2, long boz1, long boz2, long doz1, long doz2) {
    dim3 grid(N / BN, M / BM, nz1 * nz2);
    if (f32out)
      gemm_nt<true><<<grid, 256, 0, stream>>>(A, lda, Bp, ldb, D, ldd, nz2,
                                              aoz1, aoz2, boz1, boz2, doz1, doz2, K);
    else
      gemm_nt<false><<<grid, 256, 0, stream>>>(A, lda, Bp, ldb, D, ldd, nz2,
                                               aoz1, aoz2, boz1, boz2, doz1, doz2, K);
  };

  // v_eff^T (2048 x 512) = W_o (2048x2048) @ W_uvT(512x2048)^T
  gemm(wo_bf, C_, wuvT, C_, veff, NLKV_, C_, NLKV_, C_, false, 1, 1, 0, 0, 0, 0, 0, 0);
  // c_q (4096 x 1536)
  gemm(x_bf, C_, wdq_bf, C_, cq_bf, NLQ_, (int)nBT, NLQ_, C_, false, 1, 1, 0, 0, 0, 0, 0, 0);
  // c_kv (4096 x 512)
  gemm(x_bf, C_, wdkv_bf, C_, ckv, NLKV_, (int)nBT, NLKV_, C_, false, 1, 1, 0, 0, 0, 0, 0, 0);
  // c_kr (4096 x 64), fp32 for RoPE
  gemm(x_bf, C_, wkr_bf, C_, ckr, DHR_, (int)nBT, DHR_, C_, true, 1, 1, 0, 0, 0, 0, 0, 0);
  // c_qr (4096 x 1024), fp32 for RoPE
  gemm(cq_bf, NLQ_, wqr_bf, NLQ_, cqr, NH_ * DHR_, (int)nBT, NH_ * DHR_, NLQ_, true,
       1, 1, 0, 0, 0, 0, 0, 0);
  rope_k_kernel<<<(B_ * T_ * 32 + 255) / 256, 256, 0, stream>>>(ckr, cosb, sinb, kr);
  rope_q_kernel<<<(B_ * T_ * NH_ * 32 + 255) / 256, 256, 0, stream>>>(cqr, cosb, sinb, qr);
  tbf_kernel<<<(int)((nBT * NLKV_ + 255) / 256), 256, 0, stream>>>(ckv, ckvT);
  // k_eff^T per head: (512 x 1536) = W_ukT[h](512x128) @ W_uq[:,h-slice](1536x128)^T
  gemm(wukT, HS_, wuq_bf, C_, keffT, NLQ_, NLKV_, NLQ_, HS_, false,
       NH_, 1, (long)NLKV_ * HS_, 0, (long)HS_, 0, (long)NLKV_ * NLQ_, 0);
  // q_abs (b,h): (1024 x 512) = c_q[b] @ k_effT[h]^T
  gemm(cq_bf, NLQ_, keffT, NLQ_, qabs, NLKV_, T_, NLKV_, NLQ_, false,
       B_, NH_, (long)T_ * NLQ_, 0, 0, (long)NLKV_ * NLQ_,
       (long)NH_ * T_ * NLKV_, (long)T_ * NLKV_);
  // attention core
  flash_mla<<<dim3(T_ / 64, NLKV_ / 128, B_ * NH_), 128, 0, stream>>>(
      qabs, qr, ckv, kr, ckvT, ylat);
  // y (b,h): (1024 x 128) = y_lat[b,h] @ v_effT[h]^T  -> d_out fp32, ldd = C
  gemm(ylat, NLKV_, veff, NLKV_, y, C_, T_, HS_, NLKV_, true,
       B_, NH_, (long)NH_ * T_ * NLKV_, (long)T_ * NLKV_, 0, (long)HS_ * NLKV_,
       (long)T_ * C_, (long)HS_);
}
